// LayerBlock_61117384622229
// MI455X (gfx1250) — matrix operand order repaired and checked by reading, not compiled, not run on hardware
//
#include <hip/hip_runtime.h>
#include <hip/hip_bf16.h>

typedef __attribute__((ext_vector_type(16))) _Float16 v16h;
typedef __attribute__((ext_vector_type(8)))  float    v8f;

constexpr int kB  = 2;
constexpr int kC  = 128;
constexpr int kN  = 6000;
constexpr int kM  = 1500;
constexpr int kH  = 4;
constexpr int kDH = 32;

// ---------------------------------------------------------------------------
// Generic 1x1-conv GEMM:  out[b,o,n] = act( W[o,:] . X[b,:,n] + bias[o] )
// One wave computes a 32(out-ch) x 32(pos) tile: 2 A fragments x 2 B fragments
// -> 4 v_wmma per K-step, 40 loads per K-step. ALL loads are unconditional
// with clamped indices (always in-bounds); out-of-range K contributions are
// zeroed by selects after the load (TAIL only); out-of-range rows/cols only
// ever produce garbage in lanes that are never stored.
// MODE 0: X = xa; MODE 1: X = concat(xa,xb) at Csplit; MODE 2: X = xa - xb.
// Fast path (TAIL=false) requires Cin % 32 == 0 && Cout % 16 == 0.
// ---------------------------------------------------------------------------
template<int MODE, bool TAIL>
__global__ __launch_bounds__(32) void gemm_wmma(
    const float* __restrict__ W, const float* __restrict__ bias,
    const float* __restrict__ xa, const float* __restrict__ xb,
    int Csplit,
    float* __restrict__ out,
    int Cout, int Cin, int Nn,
    const float* __restrict__ scale, const float* __restrict__ shift,
    int do_relu, const float* __restrict__ residual)
{
    const int lane = threadIdx.x & 31;
    const int col = lane & 15;
    const int hi  = lane >> 4;
    const int b   = blockIdx.z;
    const int o0  = blockIdx.y * 32;
    const int n0  = blockIdx.x * 32;

    // Clamped (always in-bounds) load indices.
    int nidx[2];
    nidx[0] = min(n0 + col, Nn - 1);
    nidx[1] = min(n0 + 16 + col, Nn - 1);
    int orow[2];
    orow[0] = min(o0 + col, Cout - 1);
    orow[1] = min(o0 + 16 + col, Cout - 1);

    v8f acc[2][2];
#pragma unroll
    for (int ot = 0; ot < 2; ++ot)
#pragma unroll
        for (int nt = 0; nt < 2; ++nt)
            acc[ot][nt] = v8f{0.f,0.f,0.f,0.f,0.f,0.f,0.f,0.f};

    for (int k0 = 0; k0 < Cin; k0 += 32) {
        // ---- A fragments (2 output-channel tiles). Lane row = orow[ot];
        // halves e -> c=k0+hi*16+e.
        v16h af[2];
#pragma unroll
        for (int ot = 0; ot < 2; ++ot) {
            if (!TAIL) {
                const float* pa = W + (size_t)orow[ot] * Cin + k0 + hi * 16;
                float4 a0 = *(const float4*)(pa);
                float4 a1 = *(const float4*)(pa + 4);
                float4 a2 = *(const float4*)(pa + 8);
                float4 a3 = *(const float4*)(pa + 12);
                af[ot][0]=(_Float16)a0.x;  af[ot][1]=(_Float16)a0.y;
                af[ot][2]=(_Float16)a0.z;  af[ot][3]=(_Float16)a0.w;
                af[ot][4]=(_Float16)a1.x;  af[ot][5]=(_Float16)a1.y;
                af[ot][6]=(_Float16)a1.z;  af[ot][7]=(_Float16)a1.w;
                af[ot][8]=(_Float16)a2.x;  af[ot][9]=(_Float16)a2.y;
                af[ot][10]=(_Float16)a2.z; af[ot][11]=(_Float16)a2.w;
                af[ot][12]=(_Float16)a3.x; af[ot][13]=(_Float16)a3.y;
                af[ot][14]=(_Float16)a3.z; af[ot][15]=(_Float16)a3.w;
            } else {
                float tmp[16];
#pragma unroll
                for (int e = 0; e < 16; ++e) {
                    int cA = k0 + (hi << 4) + e;
                    tmp[e] = W[(size_t)orow[ot] * Cin + min(cA, Cin - 1)];
                    tmp[e] = (cA < Cin) ? tmp[e] : 0.0f;   // zero OOB K (branchless)
                }
#pragma unroll
                for (int e = 0; e < 16; ++e) af[ot][e] = (_Float16)tmp[e];
            }
        }

        // ---- B fragments (2 position tiles). Halves e -> c = cbase + e.
        // cbase is a multiple of 16, so a concat split (Csplit%16==0) never
        // straddles sources within one 16-run.
        const int cbase = k0 + (hi << 4);
        const float* srcA;
        const float* srcB = nullptr;
        if (MODE == 1) {
            srcA = (cbase < Csplit)
                   ? xa + ((size_t)b * Csplit + cbase) * Nn
                   : xb + ((size_t)b * (Cin - Csplit) + (cbase - Csplit)) * Nn;
        } else {
            srcA = xa + ((size_t)b * Cin + cbase) * Nn;
            if (MODE == 2) srcB = xb + ((size_t)b * Cin + cbase) * Nn;
        }

        v16h bf[2];
#pragma unroll
        for (int nt = 0; nt < 2; ++nt) {
            float tmp[16];
            if (!TAIL) {
#pragma unroll
                for (int e = 0; e < 16; ++e)
                    tmp[e] = srcA[(size_t)e * Nn + nidx[nt]];
                if (MODE == 2) {
#pragma unroll
                    for (int e = 0; e < 16; ++e)
                        tmp[e] -= srcB[(size_t)e * Nn + nidx[nt]];
                }
            } else {
#pragma unroll
                for (int e = 0; e < 16; ++e) {
                    int c = cbase + e;
                    tmp[e] = xa[((size_t)b * Cin + min(c, Cin - 1)) * Nn + nidx[nt]];
                    tmp[e] = (c < Cin) ? tmp[e] : 0.0f;    // zero OOB K (branchless)
                }
            }
#pragma unroll
            for (int e = 0; e < 16; ++e) bf[nt][e] = (_Float16)tmp[e];
        }

#pragma unroll
        for (int ot = 0; ot < 2; ++ot)
#pragma unroll
            for (int nt = 0; nt < 2; ++nt)
                acc[ot][nt] = __builtin_amdgcn_wmma_f32_16x16x32_f16(
                    false, af[ot], false, bf[nt], (short)0, acc[ot][nt], false, false);
    }

    // ---- Epilogue (guards only here, after all WMMAs).
#pragma unroll
    for (int nt = 0; nt < 2; ++nt) {
        int n = n0 + nt * 16 + col;
        if (n < Nn) {
#pragma unroll
            for (int ot = 0; ot < 2; ++ot) {
#pragma unroll
                for (int j = 0; j < 8; ++j) {
                    int o = o0 + ot * 16 + j + (hi << 3);
                    if (o < Cout) {
                        float vr = acc[ot][nt][j] + bias[o];
                        if (scale)    vr = vr * scale[o] + shift[o];
                        if (do_relu)  vr = fmaxf(vr, 0.0f);
                        if (residual) vr += residual[((size_t)b * Cout + o) * Nn + n];
                        out[((size_t)b * Cout + o) * Nn + n] = vr;
                    }
                }
            }
        }
    }
}

// ---------------------------------------------------------------------------
// Fused flash-attention core (per head, DH=32 == one WMMA K step).
// add[b, h*32+d, n] = sum_m softmax_m( mask(QK)/sqrt(32) ) * v[d,m]
// One wave per 16-query tile, KV streamed in tiles of 32 with online softmax.
// S computed transposed (rows = kv m, cols = q n): softmax reductions are
// in-lane + one xor-16 shuffle. All loads clamped & unconditional; OOB kv
// rows forced to -1e30 (exp -> 0) before softmax.
// ---------------------------------------------------------------------------
template<bool MASKED>
__global__ __launch_bounds__(32) void attn_wmma(
    const float* __restrict__ q, const float* __restrict__ k,
    const float* __restrict__ v,
    float* __restrict__ outAdd, int Nq, int Nk,
    const float* __restrict__ maskLogits /* [B,Nk]; mask = (>0) */)
{
    const int lane = threadIdx.x & 31;
    const int qt = blockIdx.x, hh = blockIdx.y, b = blockIdx.z;
    const int col = lane & 15;
    const int hi  = lane >> 4;
    const int n   = qt * 16 + col;
    const int nc  = min(n, Nq - 1);
    const float rs = 0.17677669529663687f;  // 1/sqrt(32)

    const float* qh = q + ((size_t)b * kC + hh * kDH) * Nq;
    const float* kh = k + ((size_t)b * kC + hh * kDH) * Nk;
    const float* vh = v + ((size_t)b * kC + hh * kDH) * Nk;
    const float* ml = MASKED ? (maskLogits + (size_t)b * Nk) : nullptr;

    // Q as B-operand of S^T gemm: halves e -> d = hi*16 + e
    v16h qb;
    {
        float tmp[16];
#pragma unroll
        for (int e = 0; e < 16; ++e) tmp[e] = qh[(size_t)((hi << 4) + e) * Nq + nc];
#pragma unroll
        for (int e = 0; e < 16; ++e) qb[e] = (_Float16)tmp[e];
    }

    v8f acc0 = {0.f,0.f,0.f,0.f,0.f,0.f,0.f,0.f};  // O rows d = 0..15
    v8f acc1 = {0.f,0.f,0.f,0.f,0.f,0.f,0.f,0.f};  // O rows d = 16..31
    float run_m = -1e30f, run_l = 0.0f;

    const float* vrow0 = vh + (size_t)col * Nk;         // d = col
    const float* vrow1 = vh + (size_t)(16 + col) * Nk;  // d = 16 + col

    for (int m0 = 0; m0 < Nk; m0 += 32) {
        // K^T as A-operand: rows m (two 16-row tiles), halves -> d.
        const int mc0 = min(m0 + col, Nk - 1);
        const int mc1 = min(m0 + 16 + col, Nk - 1);
        v16h ka0, ka1;
        {
            float t0[16], t1[16];
#pragma unroll
            for (int e = 0; e < 16; ++e) {
                int d = (hi << 4) + e;
                const float* kc = kh + (size_t)d * Nk;
                t0[e] = kc[mc0];
                t1[e] = kc[mc1];
            }
#pragma unroll
            for (int e = 0; e < 16; ++e) { ka0[e] = (_Float16)t0[e]; ka1[e] = (_Float16)t1[e]; }
        }
        v8f z = {0.f,0.f,0.f,0.f,0.f,0.f,0.f,0.f};
        v8f s0 = __builtin_amdgcn_wmma_f32_16x16x32_f16(false, ka0, false, qb, (short)0, z, false, false);
        v8f s1 = __builtin_amdgcn_wmma_f32_16x16x32_f16(false, ka1, false, qb, (short)0, z, false, false);

        // mask (pre-scale, like reference) + scale; OOB kv -> -1e30. Branchless.
        float mv0[8], mv1[8];
        if (MASKED) {
#pragma unroll
            for (int j = 0; j < 8; ++j) {
                int mm0 = m0 + j + (hi << 3);
                mv0[j] = ml[min(mm0, Nk - 1)];
                mv1[j] = ml[min(mm0 + 16, Nk - 1)];
            }
        }
        float p0[8], p1[8];
        float tmax = -1e30f;
#pragma unroll
        for (int j = 0; j < 8; ++j) {
            int mm0 = m0 + j + (hi << 3);         // D row for s0
            int mm1 = mm0 + 16;
            float sv0 = s0[j];
            float sv1 = s1[j];
            if (MASKED) {
                sv0 = (mv0[j] > 0.0f) ? sv0 : -1e6f;
                sv1 = (mv1[j] > 0.0f) ? sv1 : -1e6f;
            }
            sv0 *= rs;
            sv1 *= rs;
            sv0 = (mm0 < Nk) ? sv0 : -1e30f;
            sv1 = (mm1 < Nk) ? sv1 : -1e30f;
            p0[j] = sv0; p1[j] = sv1;
            tmax = fmaxf(tmax, fmaxf(sv0, sv1));
        }
        tmax = fmaxf(tmax, __shfl_xor(tmax, 16, 32));
        float new_m = fmaxf(run_m, tmax);
        float sc = __expf(run_m - new_m);
        float rsum = 0.0f;
#pragma unroll
        for (int j = 0; j < 8; ++j) {
            p0[j] = __expf(p0[j] - new_m);
            p1[j] = __expf(p1[j] - new_m);
            rsum += p0[j] + p1[j];
        }
        rsum += __shfl_xor(rsum, 16, 32);
        run_l = run_l * sc + rsum;
        run_m = new_m;
#pragma unroll
        for (int j = 0; j < 8; ++j) { acc0[j] *= sc; acc1[j] *= sc; }

        // Pack P as B-operand (rows m, cols n): halves e -> m = hi*16 + e.
        // lane<16 needs m0+8..15 from partner s0; lane>=16 needs m0+16..23
        // from partner s1 (xor-16 exchange).
        v16h pb;
#pragma unroll
        for (int j = 0; j < 8; ++j) {
            float o0 = __shfl_xor(p0[j], 16, 32);
            float o1 = __shfl_xor(p1[j], 16, 32);
            float lo = hi ? o1    : p0[j];
            float hv = hi ? p1[j] : o0;
            pb[j]     = (_Float16)lo;
            pb[j + 8] = (_Float16)hv;
        }

        // V as A-operand: A[d][m], two 16-row d tiles; clamped m (P==0 there).
        v16h va0, va1;
        {
            float t0[16], t1[16];
#pragma unroll
            for (int e = 0; e < 16; ++e) {
                int mrel = (hi << 4) + e;
                int mm = min(m0 + mrel, Nk - 1);
                t0[e] = vrow0[mm];
                t1[e] = vrow1[mm];
            }
#pragma unroll
            for (int e = 0; e < 16; ++e) { va0[e] = (_Float16)t0[e]; va1[e] = (_Float16)t1[e]; }
        }
        acc0 = __builtin_amdgcn_wmma_f32_16x16x32_f16(false, va0, false, pb, (short)0, acc0, false, false);
        acc1 = __builtin_amdgcn_wmma_f32_16x16x32_f16(false, va1, false, pb, (short)0, acc1, false, false);
    }

    float inv_l = 1.0f / run_l;
    if (n < Nq) {
#pragma unroll
        for (int j = 0; j < 8; ++j) {
            int d0 = j + (hi << 3);
            outAdd[((size_t)b * kC + hh * kDH + d0) * Nq + n]      = acc0[j] * inv_l;
            outAdd[((size_t)b * kC + hh * kDH + 16 + d0) * Nq + n] = acc1[j] * inv_l;
        }
    }
}

// ---------------------------------------------------------------------------
// In-place InstanceNorm1d(eval, eps=1e-3, biased var) + affine + ReLU over N.
// grid.x = B*Cc, one block per (b,c) row.
// ---------------------------------------------------------------------------
__global__ void instnorm_kernel(float* __restrict__ y, int Cc, int Nn,
                                const float* __restrict__ s, const float* __restrict__ sh)
{
    __shared__ float red[256];
    __shared__ float red2[256];
    const int bc = blockIdx.x;
    const int c = bc % Cc;
    float* row = y + (size_t)bc * Nn;

    float sum = 0.0f, sum2 = 0.0f;
    for (int i = threadIdx.x; i < Nn; i += blockDim.x) {
        float t = row[i]; sum += t; sum2 += t * t;
    }
    red[threadIdx.x] = sum; red2[threadIdx.x] = sum2;
    __syncthreads();
    for (int st = blockDim.x >> 1; st > 0; st >>= 1) {
        if ((int)threadIdx.x < st) {
            red[threadIdx.x]  += red[threadIdx.x + st];
            red2[threadIdx.x] += red2[threadIdx.x + st];
        }
        __syncthreads();
    }
    float mu  = red[0] / (float)Nn;
    float var = red2[0] / (float)Nn - mu * mu;
    float inv = rsqrtf(var + 1e-3f);
    float scl = s[c] * inv;
    float shv = sh[c] - mu * scl;
    for (int i = threadIdx.x; i < Nn; i += blockDim.x) {
        row[i] = fmaxf(row[i] * scl + shv, 0.0f);
    }
}

// ---------------------------------------------------------------------------
static void run_gemm(hipStream_t st, const float* W, const float* bias,
                     const float* xa, const float* xb, int mode, int Csplit,
                     float* out, int Cout, int Cin, int Nn,
                     const float* scale, const float* shift, int relu,
                     const float* resid)
{
    dim3 grid((Nn + 31) / 32, (Cout + 31) / 32, kB);
    const bool tail = (Cin % 32 != 0) || (Cout % 16 != 0);
    if (tail) {
        gemm_wmma<0, true><<<grid, dim3(32), 0, st>>>(W, bias, xa, xb, Csplit,
            out, Cout, Cin, Nn, scale, shift, relu, resid);
    } else if (mode == 1) {
        gemm_wmma<1, false><<<grid, dim3(32), 0, st>>>(W, bias, xa, xb, Csplit,
            out, Cout, Cin, Nn, scale, shift, relu, resid);
    } else if (mode == 2) {
        gemm_wmma<2, false><<<grid, dim3(32), 0, st>>>(W, bias, xa, xb, Csplit,
            out, Cout, Cin, Nn, scale, shift, relu, resid);
    } else {
        gemm_wmma<0, false><<<grid, dim3(32), 0, st>>>(W, bias, xa, xb, Csplit,
            out, Cout, Cin, Nn, scale, shift, relu, resid);
    }
}

extern "C" void kernel_launch(void* const* d_in, const int* in_sizes, int n_in,
                              void* d_out, int out_size, void* d_ws, size_t ws_size,
                              hipStream_t stream)
{
    (void)in_sizes; (void)n_in; (void)out_size; (void)ws_size;
    // Inputs in setup_inputs() order:
    const float* d_inp   = (const float*)d_in[1];
    const float* featp   = (const float*)d_in[2];
    const float* lg_in   = (const float*)d_in[3];
    const float* Wq = (const float*)d_in[4];  const float* bq = (const float*)d_in[5];
    const float* Wk = (const float*)d_in[6];  const float* bk = (const float*)d_in[7];
    const float* Wv = (const float*)d_in[8];  const float* bv = (const float*)d_in[9];
    const float* Wm = (const float*)d_in[10]; const float* bm = (const float*)d_in[11];
    const float* Wc1 = (const float*)d_in[12]; const float* bc1 = (const float*)d_in[13];
    const float* bns = (const float*)d_in[14]; const float* bnsh = (const float*)d_in[15];
    const float* Wc2 = (const float*)d_in[16]; const float* bc2 = (const float*)d_in[17];
    const float* iW1 = (const float*)d_in[18]; const float* ib1 = (const float*)d_in[19];
    const float* ibn1s = (const float*)d_in[20]; const float* ibn1sh = (const float*)d_in[21];
    const float* iW2 = (const float*)d_in[22]; const float* ib2 = (const float*)d_in[23];
    const float* ibn2s = (const float*)d_in[24]; const float* ibn2sh = (const float*)d_in[25];
    const float* iW3 = (const float*)d_in[26]; const float* ib3 = (const float*)d_in[27];
    const float* ibn3s = (const float*)d_in[28]; const float* ibn3sh = (const float*)d_in[29];
    const float* iW4 = (const float*)d_in[30]; const float* ib4 = (const float*)d_in[31];

    float* out = (float*)d_out;                 // [B,C,N] d_new, then [B,N] logits
    float* logits_out = out + (size_t)kB * kC * kN;
    float* ws = (float*)d_ws;

    // Workspace layout (floats), with reuse:
    const size_t szCN = (size_t)kB * kC * kN;   // 1,536,000
    float* buf_q = ws;                          // region A first half
    float* buf_v = ws + szCN;                   // region A second half
    float* buf_h = ws;                          // overlays A (q+v dead by then)
    float* buf_k = ws + 2 * szCN;               // region B
    float* buf_m = buf_k;                       // reuses B (k dead after attn)
    float* buf_s = ws + 3 * szCN;               // region Cr
    float* buf_fa = ws + 4 * szCN;              // feat buffers [B,C,M]
    float* buf_fb = buf_fa + (size_t)kB * kC * kM;
    float* buf_y1 = buf_s;                      // MLP overlays Cr
    float* buf_y2 = buf_y1 + (size_t)kB * 64 * kN;
    float* buf_y3 = buf_y2 + (size_t)kB * 16 * kN;

    // One attention block: out = x1 + Wc2( relu(bn( Wc1(concat(x1, Wm(attn))))) )
    auto attn_layer = [&](int i, const float* x1, int Nq, const float* x2, int Nk,
                          const float* maskp, float* outBuf) {
        const float* Wq_i = Wq + (size_t)i * kC * kC;
        const float* Wk_i = Wk + (size_t)i * kC * kC;
        const float* Wv_i = Wv + (size_t)i * kC * kC;
        const float* Wm_i = Wm + (size_t)i * kC * kC;
        const float* Wc1_i = Wc1 + (size_t)i * 2 * kC * 2 * kC;
        const float* Wc2_i = Wc2 + (size_t)i * kC * 2 * kC;

        run_gemm(stream, Wq_i, bq + (size_t)i * kC, x1, nullptr, 0, 0,
                 buf_q, kC, kC, Nq, nullptr, nullptr, 0, nullptr);
        run_gemm(stream, Wk_i, bk + (size_t)i * kC, x2, nullptr, 0, 0,
                 buf_k, kC, kC, Nk, nullptr, nullptr, 0, nullptr);
        run_gemm(stream, Wv_i, bv + (size_t)i * kC, x2, nullptr, 0, 0,
                 buf_v, kC, kC, Nk, nullptr, nullptr, 0, nullptr);

        dim3 ag((Nq + 15) / 16, kH, kB);
        if (maskp) {
            attn_wmma<true><<<ag, dim3(32), 0, stream>>>(buf_q, buf_k, buf_v, buf_s, Nq, Nk, maskp);
        } else {
            attn_wmma<false><<<ag, dim3(32), 0, stream>>>(buf_q, buf_k, buf_v, buf_s, Nq, Nk, nullptr);
        }

        run_gemm(stream, Wm_i, bm + (size_t)i * kC, buf_s, nullptr, 0, 0,
                 buf_m, kC, kC, Nq, nullptr, nullptr, 0, nullptr);
        // h = relu(bn(Wc1 . concat(x1, add)))
        run_gemm(stream, Wc1_i, bc1 + (size_t)i * 2 * kC, x1, buf_m, 1, kC,
                 buf_h, 2 * kC, 2 * kC, Nq, bns + (size_t)i * 2 * kC,
                 bnsh + (size_t)i * 2 * kC, 1, nullptr);
        // out = x1 + Wc2 . h
        run_gemm(stream, Wc2_i, bc2 + (size_t)i * kC, buf_h, nullptr, 0, 0,
                 outBuf, kC, 2 * kC, Nq, nullptr, nullptr, 0, x1);
    };

    attn_layer(0, featp, kM, d_inp, kN, lg_in, buf_fa);     // cluster (masked kv)
    attn_layer(1, buf_fa, kM, buf_fa, kM, nullptr, buf_fb); // context self-attn
    attn_layer(2, d_inp, kN, buf_fb, kM, nullptr, out);     // decluster -> d_new

    // Inlier MLP on x = d_new - d_old with instance-norm blocks
    run_gemm(stream, iW1, ib1, out, d_inp, 2, 0, buf_y1, 64, kC, kN,
             nullptr, nullptr, 0, nullptr);
    instnorm_kernel<<<dim3(kB * 64), dim3(256), 0, stream>>>(buf_y1, 64, kN, ibn1s, ibn1sh);

    run_gemm(stream, iW2, ib2, buf_y1, nullptr, 0, 0, buf_y2, 16, 64, kN,
             nullptr, nullptr, 0, nullptr);
    instnorm_kernel<<<dim3(kB * 16), dim3(256), 0, stream>>>(buf_y2, 16, kN, ibn2s, ibn2sh);

    run_gemm(stream, iW3, ib3, buf_y2, nullptr, 0, 0, buf_y3, 4, 16, kN,
             nullptr, nullptr, 0, nullptr);
    instnorm_kernel<<<dim3(kB * 4), dim3(256), 0, stream>>>(buf_y3, 4, kN, ibn3s, ibn3sh);

    // logits[b,n] = (iW4 . x)[b,0,n]  (Cout=1 -> contiguous [B,N] layout)
    run_gemm(stream, iW4, ib4, buf_y3, nullptr, 0, 0, logits_out, 1, 4, kN,
             nullptr, nullptr, 0, nullptr);
}
